// IterativeClassifier_14113262535115
// MI455X (gfx1250) — matrix operand order repaired and checked by reading, not compiled, not run on hardware
//
#include <hip/hip_runtime.h>

#define E       64
#define IN      512
#define NC      10
#define ROWS    128
#define THREADS 256   // 8 wave32 per block, 16 rows per wave

typedef __attribute__((ext_vector_type(16))) _Float16 v16h;
typedef __attribute__((ext_vector_type(8)))  _Float16 v8h;
typedef __attribute__((ext_vector_type(8)))  float    v8f;
typedef __attribute__((ext_vector_type(4)))  float    v4f;

#define WMMA(a,b,c) __builtin_amdgcn_wmma_f32_16x16x32_f16(false,(a),false,(b),(short)0,(c),false,false)
// ds_store -> ds_load ordering fence (cross-lane reuse of wave-private LDS tile);
// the memory clobber also pins in-loop LDS fragment reloads (prevents LICM hoisting).
#define LDS_FENCE() asm volatile("s_wait_dscnt 0x0" ::: "memory")

#if defined(__AMDGCN__) && __has_builtin(__builtin_amdgcn_global_load_async_to_lds_b128)
#define HAVE_ASYNC_LDS 1
// builtin signature (from hipcc diagnostic): param0 = int4* in AS1 (__device__),
// param1 = int4* in AS3 (__shared__), then imm offset, imm cpol.
typedef int v4i_gcc __attribute__((vector_size(16)));
typedef __attribute__((address_space(1))) void      GASVoid;
typedef __attribute__((address_space(3))) void      LASVoid;
typedef __attribute__((address_space(1))) v4i_gcc   GV4i;
typedef __attribute__((address_space(3))) v4i_gcc   LV4i;
#else
#define HAVE_ASYNC_LDS 0
#endif

__device__ __forceinline__ v8h cvt8(const float* p) {
  v4f a = *(const v4f*)p;
  v4f b = *(const v4f*)(p + 4);
  v8h r;
  r[0]=(_Float16)a[0]; r[1]=(_Float16)a[1]; r[2]=(_Float16)a[2]; r[3]=(_Float16)a[3];
  r[4]=(_Float16)b[0]; r[5]=(_Float16)b[1]; r[6]=(_Float16)b[2]; r[7]=(_Float16)b[3];
  return r;
}
// non-temporal variant for the once-streamed x rows
__device__ __forceinline__ v8h cvt8_nt(const float* p) {
  v4f a = __builtin_nontemporal_load((const v4f*)p);
  v4f b = __builtin_nontemporal_load((const v4f*)(p + 4));
  v8h r;
  r[0]=(_Float16)a[0]; r[1]=(_Float16)a[1]; r[2]=(_Float16)a[2]; r[3]=(_Float16)a[3];
  r[4]=(_Float16)b[0]; r[5]=(_Float16)b[1]; r[6]=(_Float16)b[2]; r[7]=(_Float16)b[3];
  return r;
}

__device__ __forceinline__ v16h pack16(v8h lo, v8h hi) {
  return __builtin_shufflevector(lo, hi, 0,1,2,3,4,5,6,7,8,9,10,11,12,13,14,15);
}

// A-frag (16x32 f16): lane L: row=L&15, g=L>>4;
// halves j = row[k0+g*16+j]
__device__ __forceinline__ v16h afrag_lds(const _Float16* rowp, int k0, int g) {
  const _Float16* p = rowp + k0 + g*16;
  return pack16(*(const v8h*)p, *(const v8h*)(p + 8));
}
__device__ __forceinline__ v16h afrag_glb_nt(const float* rowp, int k0, int g) {
  const float* p = rowp + k0 + g*16;
  return pack16(cvt8_nt(p), cvt8_nt(p + 8));
}
// B-frag (32x16 f16): lane L: col=L&15, g=L>>4; halves j = W[col][k0+g*16+j]
__device__ __forceinline__ v16h bfrag_lds(const _Float16* p) {
  return pack16(*(const v8h*)p, *(const v8h*)(p + 8));
}

__global__ __launch_bounds__(THREADS, 1)
void itcls_kernel(const float* __restrict__ x,     const float* __restrict__ z0,
                  const float* __restrict__ Wfeat, const float* __restrict__ bfeat,
                  const float* __restrict__ W1,    const float* __restrict__ b1,
                  const float* __restrict__ W2,    const float* __restrict__ b2,
                  const float* __restrict__ CE,    const int*   __restrict__ Tp,
                  float* __restrict__ out)
{
  extern __shared__ char smem[];
  _Float16* sWf    = (_Float16*)smem;      // [64][512] f16
  _Float16* sW1f   = sWf  + E*IN;          // [64][64]  W1[:, 0:64]
  _Float16* sW1z   = sW1f + E*E;           // [64][64]  W1[:, 64:128]
  _Float16* sW2    = sW1z + E*E;           // [64][64]
  _Float16* sScr   = sW2  + E*E;           // 8 waves * [16][72]
  float*    sF     = (float*)(sScr + 8*16*72);
  float*    sw1t   = sF;                   // W1[:,128]
  float*    sb1    = sF + E;
  float*    sb2    = sF + 2*E;
  float*    sbf    = sF + 3*E;
  float*    sStage = sF + 4*E;             // f32 staging buffer [8192] (32KB)

  const int tid = threadIdx.x;

  // ---- stage small weights to LDS as f16 (once per block)
  for (int i = tid; i < E*E; i += THREADS) {
    int n = i >> 6, k = i & 63;
    sW1f[i] = (_Float16)W1[n*(2*E+1) + k];
    sW1z[i] = (_Float16)W1[n*(2*E+1) + E + k];
    sW2[i]  = (_Float16)W2[i];
  }
  if (tid < E) {
    sw1t[tid] = W1[tid*(2*E+1) + 2*E];
    sb1[tid]  = b1[tid];
    sb2[tid]  = b2[tid];
    sbf[tid]  = bfeat[tid];
  }

  // ---- stage W_feat (128KB f32 -> 64KB f16) via async global->LDS DMA if available
#if HAVE_ASYNC_LDS
  for (int chunk = 0; chunk < 4; ++chunk) {
    #pragma unroll
    for (int i = 0; i < 8; ++i) {
      const int f = (i*THREADS + tid) * 4;             // 16B per thread per issue
      __builtin_amdgcn_global_load_async_to_lds_b128(
          (GV4i*)(GASVoid*)(Wfeat + chunk*8192 + f),
          (LV4i*)(LASVoid*)(sStage + f), 0, 0);
    }
    asm volatile("s_wait_asynccnt 0x0" ::: "memory");
    __syncthreads();
    for (int i = tid; i < 8192; i += THREADS)
      sWf[chunk*8192 + i] = (_Float16)sStage[i];
    __syncthreads();
  }
#else
  for (int i = tid; i < E*IN; i += THREADS) sWf[i] = (_Float16)Wfeat[i];
#endif
  __syncthreads();

  const int T    = Tp[0];
  const int wave = tid >> 5;
  const int lane = tid & 31;
  const int c    = lane & 15;
  const int g    = lane >> 4;
  const int rowbase = blockIdx.x * ROWS + wave * 16;
  _Float16* scr = sScr + wave * (16*72);

  // per-lane bias scalars for n = nt*16 + c
  float b1v[4], w1tv[4], b2v[4], bfv[4];
  #pragma unroll
  for (int nt = 0; nt < 4; ++nt) {
    b1v[nt]  = sb1[nt*16 + c];
    w1tv[nt] = sw1t[nt*16 + c];
    b2v[nt]  = sb2[nt*16 + c];
    bfv[nt]  = sbf[nt*16 + c];
  }

  // ---- features = x @ Wfeat^T + bfeat   (K = 512 = 16 WMMA k-steps)
  const float* xrow = x + (size_t)(rowbase + c) * IN;
  v8f facc[4] = {};
  #pragma unroll 4
  for (int kc = 0; kc < IN/32; ++kc) {
    const int pfo = (kc + 4 < IN/32) ? (kc + 4)*32 : 0;   // scalar select, no branch
    __builtin_prefetch(xrow + pfo, 0, 3);
    v16h xa = afrag_glb_nt(xrow, kc*32, g);
    #pragma unroll
    for (int nt = 0; nt < 4; ++nt) {
      v16h wb = bfrag_lds(sWf + (nt*16 + c)*IN + kc*32 + g*16);
      facc[nt] = WMMA(xa, wb, facc[nt]);
    }
  }
  // features (+bias) -> wave-private LDS tile -> A-fragments (held for all steps)
  #pragma unroll
  for (int nt = 0; nt < 4; ++nt)
    #pragma unroll
    for (int v = 0; v < 8; ++v)
      scr[(v + g*8)*72 + nt*16 + c] = (_Float16)(facc[nt][v] + bfv[nt]);
  LDS_FENCE();
  v16h fa0 = afrag_lds(scr + c*72, 0, g);
  v16h fa1 = afrag_lds(scr + c*72, 32, g);
  LDS_FENCE();

  // ---- W1 fragments register-resident across the loop (loaded after feature
  // phase so they don't overlap facc live ranges); W2 is reloaded per step.
  v16h w1fF[2][4], w1zF[2][4];
  #pragma unroll
  for (int kc = 0; kc < 2; ++kc)
    #pragma unroll
    for (int nt = 0; nt < 4; ++nt) {
      const int off = (nt*16 + c)*E + kc*32 + g*16;
      w1fF[kc][nt] = bfrag_lds(sW1f + off);
      w1zF[kc][nt] = bfrag_lds(sW1z + off);
    }

  // ---- z state in f32, C-fragment layout (m = v + g*8, n = nt*16 + c)
  v8f zc[4];
  #pragma unroll
  for (int nt = 0; nt < 4; ++nt)
    #pragma unroll
    for (int v = 0; v < 8; ++v)
      zc[nt][v] = __builtin_nontemporal_load(
                    z0 + (size_t)(rowbase + v + g*8)*E + nt*16 + c);

  // ---- time loop: 24 WMMA per wave per step, everything on-chip
  for (int s = 0; s < T; ++s) {
    float t = (float)s / (float)T;

    #pragma unroll
    for (int nt = 0; nt < 4; ++nt)
      #pragma unroll
      for (int v = 0; v < 8; ++v)
        scr[(v + g*8)*72 + nt*16 + c] = (_Float16)zc[nt][v];
    LDS_FENCE();
    v16h za0 = afrag_lds(scr + c*72, 0, g);
    v16h za1 = afrag_lds(scr + c*72, 32, g);
    LDS_FENCE();

    v8f acc[4];
    #pragma unroll
    for (int nt = 0; nt < 4; ++nt) {
      v8f a = {};
      a = WMMA(fa0, w1fF[0][nt], a);
      a = WMMA(fa1, w1fF[1][nt], a);
      a = WMMA(za0, w1zF[0][nt], a);
      a = WMMA(za1, w1zF[1][nt], a);
      acc[nt] = a;
    }

    // h = relu(acc + b1 + t*w1t) -> LDS tile -> A-frags
    #pragma unroll
    for (int nt = 0; nt < 4; ++nt) {
      float bb = b1v[nt] + t * w1tv[nt];
      #pragma unroll
      for (int v = 0; v < 8; ++v) {
        float h = fmaxf(acc[nt][v] + bb, 0.0f);
        scr[(v + g*8)*72 + nt*16 + c] = (_Float16)h;
      }
    }
    LDS_FENCE();
    v16h ha0 = afrag_lds(scr + c*72, 0, g);
    v16h ha1 = afrag_lds(scr + c*72, 32, g);
    LDS_FENCE();

    // W2 fragments reloaded from LDS each step (kept off the register budget;
    // the fence memory-clobber above prevents these from being hoisted).
    #pragma unroll
    for (int nt = 0; nt < 4; ++nt) {
      const int off = (nt*16 + c)*E + g*16;
      v16h w2a = bfrag_lds(sW2 + off);
      v16h w2b = bfrag_lds(sW2 + off + 32);
      v8f a = {};
      a = WMMA(ha0, w2a, a);
      a = WMMA(ha1, w2b, a);
      #pragma unroll
      for (int v = 0; v < 8; ++v) {
        float pred = a[v] + b2v[nt];
        zc[nt][v] += 0.1f * (pred - zc[nt][v]);
      }
    }
  }

  // ---- logits = z @ class_emb^T (pad N=10 -> 16)
  #pragma unroll
  for (int nt = 0; nt < 4; ++nt)
    #pragma unroll
    for (int v = 0; v < 8; ++v)
      scr[(v + g*8)*72 + nt*16 + c] = (_Float16)zc[nt][v];
  LDS_FENCE();
  v16h za0 = afrag_lds(scr + c*72, 0, g);
  v16h za1 = afrag_lds(scr + c*72, 32, g);
  LDS_FENCE();

  v16h ce0, ce1;
  #pragma unroll
  for (int i = 0; i < 16; ++i) { ce0[i] = (_Float16)0.0f; ce1[i] = (_Float16)0.0f; }
  if (c < NC) {
    const float* cp = CE + c*E;
    ce0 = pack16(cvt8(cp + g*16),      cvt8(cp + g*16 + 8));
    ce1 = pack16(cvt8(cp + 32 + g*16), cvt8(cp + 32 + g*16 + 8));
  }
  v8f lacc = {};
  lacc = WMMA(za0, ce0, lacc);
  lacc = WMMA(za1, ce1, lacc);

  if (c < NC) {
    #pragma unroll
    for (int v = 0; v < 8; ++v)
      __builtin_nontemporal_store(lacc[v],
          out + (size_t)(rowbase + v + g*8)*NC + c);
  }
}

extern "C" void kernel_launch(void* const* d_in, const int* in_sizes, int n_in,
                              void* d_out, int out_size, void* d_ws, size_t ws_size,
                              hipStream_t stream) {
  const float* x     = (const float*)d_in[0];
  const float* z0    = (const float*)d_in[1];
  const float* Wfeat = (const float*)d_in[2];
  const float* bfeat = (const float*)d_in[3];
  const float* W1    = (const float*)d_in[4];
  const float* b1    = (const float*)d_in[5];
  const float* W2    = (const float*)d_in[6];
  const float* b2    = (const float*)d_in[7];
  const float* CE    = (const float*)d_in[8];
  const int*   T     = (const int*)d_in[9];
  float* out = (float*)d_out;

  const int nrows  = in_sizes[0] / IN;     // 65536
  const int blocks = nrows / ROWS;         // 512
  const size_t lds = (size_t)(E*IN + 3*E*E + 8*16*72) * sizeof(_Float16)
                   + (size_t)(4*E) * sizeof(float)
                   + (size_t)8192 * sizeof(float);      // ~139 KB dynamic LDS
  (void)hipFuncSetAttribute((const void*)itcls_kernel,
                            hipFuncAttributeMaxDynamicSharedMemorySize, (int)lds);
  itcls_kernel<<<blocks, THREADS, lds, stream>>>(x, z0, Wfeat, bfeat, W1, b1,
                                                 W2, b2, CE, T, out);
}
